// MultiHeadAttentionLayer_28295244546584
// MI455X (gfx1250) — matrix operand order repaired and checked by reading, not compiled, not run on hardware
//
#include <hip/hip_runtime.h>

#define DEVINL __device__ __forceinline__

constexpr int kNodeDim = 128;
constexpr int kEdgeDim = 32;
constexpr int kHeads   = 8;
constexpr int kE       = 250000;

typedef __attribute__((ext_vector_type(16))) __bf16    v16bf;
typedef __attribute__((ext_vector_type(8)))  float     v8f;
typedef __attribute__((ext_vector_type(8)))  unsigned  v8u;

DEVINL __bf16 f2bf(float f) { return (__bf16)f; }
DEVINL float bf2f(unsigned short h) {
  unsigned u = ((unsigned)h) << 16;
  return __builtin_bit_cast(float, u);
}
DEVINL unsigned short bfbits(float f) {
  return __builtin_bit_cast(unsigned short, (__bf16)f);
}

DEVINL v8f wmma_bf16(v16bf a, v16bf b, v8f c) {
  return __builtin_amdgcn_wmma_f32_16x16x32_bf16(false, a, false, b, (short)0, c,
                                                 false, false);
}

// A fragment (16x32 bf16). p = &src[row*ld + kchunk + half*8], 16B aligned.
// idx 0..7 -> K = base+idx, idx 8..15 -> K = base+16+idx-8 (ISA 16-bit A layout).
DEVINL v16bf load_a_frag(const float* __restrict__ p) {
  const float4* q = (const float4*)p;
  const float4 a0 = q[0], a1 = q[1];  // K = base+0..7
  const float4 b0 = q[4], b1 = q[5];  // K = base+16..23
  v16bf a;
  a[0]  = f2bf(a0.x); a[1]  = f2bf(a0.y); a[2]  = f2bf(a0.z); a[3]  = f2bf(a0.w);
  a[4]  = f2bf(a1.x); a[5]  = f2bf(a1.y); a[6]  = f2bf(a1.z); a[7]  = f2bf(a1.w);
  a[8]  = f2bf(b0.x); a[9]  = f2bf(b0.y); a[10] = f2bf(b0.z); a[11] = f2bf(b0.w);
  a[12] = f2bf(b1.x); a[13] = f2bf(b1.y); a[14] = f2bf(b1.z); a[15] = f2bf(b1.w);
  return a;
}

// Pre-packed B fragment: per tile, 32 lanes x 16 bf16 contiguous (32B per lane)
// -> two global_load_b128 per fragment.
DEVINL v16bf load_pb(const unsigned short* __restrict__ pw, int tile, int lane) {
  const uint4* p = (const uint4*)(pw + ((size_t)(tile * 32 + lane)) * 16);
  const uint4 q0 = p[0], q1 = p[1];
  v8u u;
  u[0] = q0.x; u[1] = q0.y; u[2] = q0.z; u[3] = q0.w;
  u[4] = q1.x; u[5] = q1.y; u[6] = q1.z; u[7] = q1.w;
  return __builtin_bit_cast(v16bf, u);
}

// order-preserving float <-> uint key for atomicMax on floats
DEVINL unsigned fkey(float f) {
  unsigned u = __builtin_bit_cast(unsigned, f);
  return (u & 0x80000000u) ? ~u : (u | 0x80000000u);
}
DEVINL float fdecode(unsigned k) {
  unsigned u = (k & 0x80000000u) ? (k ^ 0x80000000u) : ~k;
  return __builtin_bit_cast(float, u);
}

__global__ void k0_init(unsigned* mkey, float* sums) {
  int i = threadIdx.x;
  if (i < kHeads) {
    mkey[i] = fkey(-3.402823466e38f);
    sums[i] = 0.f;
  }
}

// Pack row-major W[kdim][128] fp32 into bf16 B-fragment layout.
// Block b: kk = b>>3, h = b&7 (tile = kk*8+h). Lane l holds col n = h*16+(l&15),
// elements j=0..7 -> K = kk*32 + (l>>4)*8 + j, j=8..15 -> K = kk*32 + 16 + (l>>4)*8 + j-8.
__global__ __launch_bounds__(32) void k_pack(const float* __restrict__ w,
                                             unsigned short* __restrict__ dst) {
  const int b = blockIdx.x;
  const int kk = b >> 3, h = b & 7;
  const int lane = threadIdx.x;
  const int half = lane >> 4;
  const int n = h * 16 + (lane & 15);
  const float* p = w + (kk * 32 + half * 8) * kNodeDim + n;
  unsigned short* d = dst + ((size_t)(b * 32 + lane)) * 16;
#pragma unroll
  for (int j = 0; j < 16; ++j) d[j] = bfbits(p[(j + (j & 8)) * kNodeDim]);
}

__global__ __launch_bounds__(256) void k1_qkv_scores(
    const float* __restrict__ xi, const float* __restrict__ xj,
    const float* __restrict__ ea, const float* __restrict__ bq,
    const float* __restrict__ bk, const float* __restrict__ bv,
    const float* __restrict__ be, const unsigned short* __restrict__ pwq,
    const unsigned short* __restrict__ pwk,
    const unsigned short* __restrict__ pwv,
    const unsigned short* __restrict__ pwe, float* __restrict__ scores,
    unsigned short* __restrict__ vstore) {
  const int lane = threadIdx.x & 31;
  const int wave = threadIdx.x >> 5;
  const int half = lane >> 4;
  const int m    = lane & 15;
  const int r0   = blockIdx.x * 128 + wave * 16;
  if (r0 >= kE) return;                // uniform per wave: EXEC stays full
  const int ra = min(r0 + m, kE - 1);  // clamped A-row for ragged tail

  const float* xir = xi + ra * kNodeDim;
  const float* xjr = xj + ra * kNodeDim;
  const float* ear = ea + ra * kEdgeDim;

  // Hoisted A fragments (head-independent): 9 x v16bf
  v16bf Ai[4], Aj[4];
#pragma unroll
  for (int kk = 0; kk < 4; ++kk) {
    Ai[kk] = load_a_frag(xir + kk * 32 + half * 8);
    Aj[kk] = load_a_frag(xjr + kk * 32 + half * 8);
  }
  const v16bf Ae = load_a_frag(ear + half * 8);

  for (int h = 0; h < kHeads; ++h) {
    const int nn = h * 16;
    v8f qacc, kacc, vacc, bacc;
    const float bqv = bq[nn + m], bkv = bk[nn + m];
    const float bvv = bv[nn + m], bev = be[nn + m];
#pragma unroll
    for (int p = 0; p < 8; ++p) {
      qacc[p] = bqv; kacc[p] = bkv; vacc[p] = bvv; bacc[p] = bev;
    }

#pragma unroll
    for (int kk = 0; kk < 4; ++kk) {
      const int tile = kk * 8 + h;
      qacc = wmma_bf16(Ai[kk], load_pb(pwq, tile, lane), qacc);
      kacc = wmma_bf16(Aj[kk], load_pb(pwk, tile, lane), kacc);
      vacc = wmma_bf16(Aj[kk], load_pb(pwv, tile, lane), vacc);
    }
    bacc = wmma_bf16(Ae, load_pb(pwe, h, lane), bacc);

    // score[row][h] = (1/sqrt(16)) * sum_N q*(k+bias); N = lane&15 in C layout
#pragma unroll
    for (int p = 0; p < 8; ++p) {
      float t = qacc[p] * (kacc[p] + bacc[p]);
      t += __shfl_xor(t, 1, 32);
      t += __shfl_xor(t, 2, 32);
      t += __shfl_xor(t, 4, 32);
      t += __shfl_xor(t, 8, 32);
      const int row = r0 + p + half * 8;
      if (m == 0 && row < kE) scores[h * kE + row] = t * 0.25f;
    }
    // V tile -> bf16 workspace (C layout: row = p + 8*half, col = nn+m)
#pragma unroll
    for (int p = 0; p < 8; ++p) {
      const int row = r0 + p + half * 8;
      if (row < kE) vstore[row * kNodeDim + nn + m] = bfbits(vacc[p]);
    }
  }
}

__global__ __launch_bounds__(256) void k2_max(const float* __restrict__ scores,
                                              unsigned* __restrict__ mkey) {
  const int h = blockIdx.y;
  const float* s = scores + h * kE;
  float lm = -3.402823466e38f;
  for (int i = blockIdx.x * blockDim.x + threadIdx.x; i < kE;
       i += gridDim.x * blockDim.x)
    lm = fmaxf(lm, s[i]);
  __shared__ float red[256];
  red[threadIdx.x] = lm;
  __syncthreads();
  for (int off = 128; off > 0; off >>= 1) {
    if (threadIdx.x < off)
      red[threadIdx.x] = fmaxf(red[threadIdx.x], red[threadIdx.x + off]);
    __syncthreads();
  }
  if (threadIdx.x == 0) atomicMax(&mkey[h], fkey(red[0]));
}

__global__ __launch_bounds__(256) void k2_sumexp(
    const float* __restrict__ scores, const unsigned* __restrict__ mkey,
    float* __restrict__ sums) {
  const int h = blockIdx.y;
  const float mh = fdecode(mkey[h]);
  const float* s = scores + h * kE;
  float ls = 0.f;
  for (int i = blockIdx.x * blockDim.x + threadIdx.x; i < kE;
       i += gridDim.x * blockDim.x)
    ls += __expf(s[i] - mh);
  __shared__ float red[256];
  red[threadIdx.x] = ls;
  __syncthreads();
  for (int off = 128; off > 0; off >>= 1) {
    if (threadIdx.x < off) red[threadIdx.x] += red[threadIdx.x + off];
    __syncthreads();
  }
  if (threadIdx.x == 0) atomicAdd(&sums[h], red[0]);
}

__global__ __launch_bounds__(256) void k3_out(
    const unsigned short* __restrict__ vstore, const float* __restrict__ scores,
    const unsigned* __restrict__ mkey, const float* __restrict__ sums,
    const unsigned short* __restrict__ pwo, const float* __restrict__ bo,
    float* __restrict__ out) {
  const int lane = threadIdx.x & 31;
  const int wave = threadIdx.x >> 5;
  const int half = lane >> 4;
  const int m    = lane & 15;
  const int r0   = blockIdx.x * 128 + wave * 16;
  if (r0 >= kE) return;
  const int ra = min(r0 + m, kE - 1);

  float wgt[kHeads];
#pragma unroll
  for (int h = 0; h < kHeads; ++h) {
    const float mh = fdecode(mkey[h]);
    const float rs = 1.0f / sums[h];
    wgt[h] = __expf(scores[h * kE + ra] - mh) * rs;
  }

  // A fragments = softmax-weighted V rows; each 8-element K group = one head.
  const unsigned short* vr = vstore + ra * kNodeDim;
  v16bf A[4];
#pragma unroll
  for (int kk = 0; kk < 4; ++kk) {
    const int c0 = kk * 32 + half * 8;
    const float w0 = wgt[c0 >> 4];
    const float w1 = wgt[(c0 + 16) >> 4];
    const uint4* q = (const uint4*)(vr + c0);
    const uint4 g0 = q[0];  // cols c0..c0+7   (8 bf16)
    const uint4 g1 = q[2];  // cols c0+16..+23 (8 bf16)
    const unsigned u0[4] = {g0.x, g0.y, g0.z, g0.w};
    const unsigned u1[4] = {g1.x, g1.y, g1.z, g1.w};
#pragma unroll
    for (int j = 0; j < 4; ++j) {
      A[kk][2 * j]     = f2bf(bf2f((unsigned short)(u0[j] & 0xFFFFu)) * w0);
      A[kk][2 * j + 1] = f2bf(bf2f((unsigned short)(u0[j] >> 16)) * w0);
      A[kk][2 * j + 8] = f2bf(bf2f((unsigned short)(u1[j] & 0xFFFFu)) * w1);
      A[kk][2 * j + 9] = f2bf(bf2f((unsigned short)(u1[j] >> 16)) * w1);
    }
  }

  for (int h = 0; h < kHeads; ++h) {
    const int nn = h * 16;
    v8f acc;
    const float bov = bo[nn + m];
#pragma unroll
    for (int p = 0; p < 8; ++p) acc[p] = bov;
#pragma unroll
    for (int kk = 0; kk < 4; ++kk)
      acc = wmma_bf16(A[kk], load_pb(pwo, kk * 8 + h, lane), acc);
#pragma unroll
    for (int p = 0; p < 8; ++p) {
      const int row = r0 + p + half * 8;
      if (row < kE) out[row * kNodeDim + nn + m] = acc[p];
    }
  }
}

extern "C" void kernel_launch(void* const* d_in, const int* in_sizes, int n_in,
                              void* d_out, int out_size, void* d_ws,
                              size_t ws_size, hipStream_t stream) {
  const float* xi = (const float*)d_in[0];
  const float* xj = (const float*)d_in[1];
  const float* ea = (const float*)d_in[2];
  const float* wq = (const float*)d_in[3];
  const float* bq = (const float*)d_in[4];
  const float* wk = (const float*)d_in[5];
  const float* bk = (const float*)d_in[6];
  const float* wv = (const float*)d_in[7];
  const float* bv = (const float*)d_in[8];
  const float* we = (const float*)d_in[9];
  const float* be = (const float*)d_in[10];
  const float* wo = (const float*)d_in[11];
  const float* bo = (const float*)d_in[12];
  float* out = (float*)d_out;

  char* ws = (char*)d_ws;
  unsigned* mkey = (unsigned*)ws;      // 8 * 4 B
  float* sums    = (float*)(ws + 64);  // 8 * 4 B
  float* scores  = (float*)(ws + 256); // [8][E] f32 = 8 MB
  size_t off = 256 + (size_t)kHeads * kE * sizeof(float);
  unsigned short* vbf = (unsigned short*)(ws + off);  // [E][128] bf16 = 64 MB
  off += (size_t)kE * kNodeDim * sizeof(unsigned short);
  // packed bf16 weight fragments: 32 tiles * 32 lanes * 16 elems (node GEMMs)
  constexpr size_t kPackN = 32u * 32u * 16u;  // per 128x128 matrix
  constexpr size_t kPackE = 8u * 32u * 16u;   // for 32x128 matrix
  unsigned short* pwq = (unsigned short*)(ws + off); off += kPackN * 2;
  unsigned short* pwk = (unsigned short*)(ws + off); off += kPackN * 2;
  unsigned short* pwv = (unsigned short*)(ws + off); off += kPackN * 2;
  unsigned short* pwo = (unsigned short*)(ws + off); off += kPackN * 2;
  unsigned short* pwe = (unsigned short*)(ws + off); off += kPackE * 2;

  const int nb = (kE + 127) / 128;  // 8 waves/block * 16 rows/wave
  hipLaunchKernelGGL(k0_init, dim3(1), dim3(32), 0, stream, mkey, sums);
  hipLaunchKernelGGL(k_pack, dim3(32), dim3(32), 0, stream, wq, pwq);
  hipLaunchKernelGGL(k_pack, dim3(32), dim3(32), 0, stream, wk, pwk);
  hipLaunchKernelGGL(k_pack, dim3(32), dim3(32), 0, stream, wv, pwv);
  hipLaunchKernelGGL(k_pack, dim3(32), dim3(32), 0, stream, wo, pwo);
  hipLaunchKernelGGL(k_pack, dim3(8), dim3(32), 0, stream, we, pwe);
  hipLaunchKernelGGL(k1_qkv_scores, dim3(nb), dim3(256), 0, stream, xi, xj, ea,
                     bq, bk, bv, be, pwq, pwk, pwv, pwe, scores, vbf);
  hipLaunchKernelGGL(k2_max, dim3(64, 8), dim3(256), 0, stream, scores, mkey);
  hipLaunchKernelGGL(k2_sumexp, dim3(64, 8), dim3(256), 0, stream, scores, mkey,
                     sums);
  hipLaunchKernelGGL(k3_out, dim3(nb), dim3(256), 0, stream, vbf, scores, mkey,
                     sums, pwo, bo, out);
}
